// FBRNN_91259465105786
// MI455X (gfx1250) — matrix operand order repaired and checked by reading, not compiled, not run on hardware
//
#include <hip/hip_runtime.h>
#include <hip/hip_bf16.h>

typedef __bf16 v16bf __attribute__((ext_vector_type(16)));
typedef float  v8f   __attribute__((ext_vector_type(8)));

// ---------------------------------------------------------------------------
// Kernel 1: convert W_ih0 (48 x 2048, f32) to bf16 in scratch.
// ---------------------------------------------------------------------------
__global__ void fbrnn_cvt_w_bf16(const float* __restrict__ w,
                                 __bf16* __restrict__ o, int n) {
    int i = blockIdx.x * blockDim.x + threadIdx.x;
    if (i < n) o[i] = (__bf16)w[i];
}

// ---------------------------------------------------------------------------
// Kernel 2: Gi0[T,48] = batch[T,2048] @ W_ih0^T + b_ih0  via bf16 WMMA.
// One wave computes a 64(M) x 48(N) supertile: 4 A-frags x 3 B-frags,
// 12 v_wmma_f32_16x16x32_bf16 per K-step, K = 2048 (64 steps).
// Fragment layouts used here (contiguous k, same order in A and B):
//   A (16x32 bf16): lane L -> m = L&15; khalf = (L>>4)*16;
//                   elems 0..7  = A[m][kk+khalf+0..7]
//                   elems 8..15 = A[m][kk+khalf+8..15]
//   B (32x16 bf16): lane L -> n = L&15; kq = (L>>4)*16;
//                   elems j = B[kk+kq+j][n] = W[n][kk+kq+j]  (contiguous!)
//   C/D (16x16 f32): lane L -> n = L&15; vgpr r -> m = r + 8*(L>>4)
// ---------------------------------------------------------------------------
__global__ __launch_bounds__(256) void fbrnn_gemm_gi0(
    const float*  __restrict__ batch,   // (16384, 2048) f32
    const __bf16* __restrict__ wbf,     // (48, 2048) bf16
    const float*  __restrict__ bias,    // (48,)
    float*        __restrict__ gi0)     // (16384, 48)
{
    const int lane  = threadIdx.x & 31;
    const int wave  = threadIdx.x >> 5;
    const int super = blockIdx.x * 8 + wave;     // 256 supertiles of 64 rows
    const int row0  = super * 64;

    const int m      = lane & 15;
    const int khalfA = (lane >> 4) * 16;
    const int n      = lane & 15;
    const int kqB    = (lane >> 4) * 16;

    v8f acc[4][3] = {};

    for (int kk = 0; kk < 2048; kk += 32) {
        v16bf bfrag[3];
#pragma unroll
        for (int g = 0; g < 3; ++g) {
            const __bf16* p = wbf + (size_t)(g * 16 + n) * 2048 + kk + kqB;
            bfrag[g] = *(const v16bf*)p;          // 32B contiguous load
        }
#pragma unroll
        for (int mt = 0; mt < 4; ++mt) {
            const float* arow =
                batch + (size_t)(row0 + mt * 16 + m) * 2048 + kk + khalfA;
            v16bf afrag;
#pragma unroll
            for (int j = 0; j < 8; ++j) afrag[j]     = (__bf16)arow[j];
#pragma unroll
            for (int j = 0; j < 8; ++j) afrag[8 + j] = (__bf16)arow[8 + j];
#pragma unroll
            for (int g = 0; g < 3; ++g)
                acc[mt][g] = __builtin_amdgcn_wmma_f32_16x16x32_bf16(
                    false, afrag, false, bfrag[g], (short)0, acc[mt][g],
                    false, false);
        }
    }

#pragma unroll
    for (int mt = 0; mt < 4; ++mt) {
#pragma unroll
        for (int g = 0; g < 3; ++g) {
            const int col = g * 16 + n;
            const float b = bias[col];
#pragma unroll
            for (int r = 0; r < 8; ++r) {
                const int rowm = r + 8 * (lane >> 4);
                gi0[(size_t)(row0 + mt * 16 + rowm) * 48 + col] =
                    acc[mt][g][r] + b;
            }
        }
    }
}

// ---------------------------------------------------------------------------
// Kernel 3: the sequential scan. Single wave32 workgroup: barriers are S_NOP
// for single-wave workgroups (CDNA5 ISA 3.1), so the whole step pipeline is
// wave-synchronous with zero sync overhead.
// ---------------------------------------------------------------------------
__device__ __forceinline__ float fbrnn_sigmoid(float x) {
    return 1.0f / (1.0f + __expf(-x));
}

__global__ __launch_bounds__(32) void fbrnn_scan(
    const float* __restrict__ gi0,   // (T,48) precomputed layer-0 gi (+bias)
    const float* __restrict__ h0,    // (6,16)
    const float* __restrict__ Whh0,  // (48,16)
    const float* __restrict__ bhh0,  // (48,)
    const float* __restrict__ Wih,   // (5,48,16)
    const float* __restrict__ Whh,   // (5,48,16)
    const float* __restrict__ bih,   // (5,48)
    const float* __restrict__ bhh,   // (5,48)
    const float* __restrict__ Wa,    // (6,16,64)
    const float* __restrict__ ba,    // (6,64)
    const float* __restrict__ va,    // (6,64)
    const float* __restrict__ fc1w,  // (32,16)
    const float* __restrict__ fc1b,  // (32,)
    const float* __restrict__ fc2w,  // (1,32)
    const float* __restrict__ fc2b,  // (1,)
    float*       __restrict__ out,   // (T,)
    int T)
{
    // GRU weights padded to stride 17 (48x16 -> 48x17) to avoid LDS bank
    // conflicts when 32 lanes each read a distinct row.
    __shared__ float sW[816 + 5 * 1632];   // Whh0 | per-layer [Wih|Whh]
    __shared__ float sWa[6144];            // (6,16,64), a-major contiguous
    __shared__ float sB[48 + 5 * 96];      // bhh0 | per-layer [bih|bhh]
    __shared__ float sH[96];               // carried hidden states (6,16)
    __shared__ float sNew[96];             // raw GRU outputs this step
    __shared__ float sGi[48], sGh[48];
    __shared__ float sInp[16];

    const int lane = threadIdx.x;

    // ---- one-time cooperative load of reused weights into LDS ----
    for (int i = lane; i < 768; i += 32)
        sW[(i >> 4) * 17 + (i & 15)] = Whh0[i];
    for (int l = 0; l < 5; ++l)
        for (int i = lane; i < 768; i += 32) {
            sW[816 + l * 1632 + (i >> 4) * 17 + (i & 15)]       = Wih[l * 768 + i];
            sW[816 + l * 1632 + 816 + (i >> 4) * 17 + (i & 15)] = Whh[l * 768 + i];
        }
    for (int i = lane; i < 6144; i += 32) sWa[i] = Wa[i];
    for (int i = lane; i < 48; i += 32)   sB[i]  = bhh0[i];
    for (int l = 0; l < 5; ++l)
        for (int i = lane; i < 48; i += 32) {
            sB[48 + l * 96 + i]      = bih[l * 48 + i];
            sB[48 + l * 96 + 48 + i] = bhh[l * 48 + i];
        }
    for (int i = lane; i < 96; i += 32) sH[i] = h0[i];
    __syncthreads();

    for (int t = 0; t < T; ++t) {
        // -------- layer 0: gi precomputed; gh = h0 @ Whh0^T + bhh0 --------
        for (int g = lane; g < 48; g += 32) {
            sGi[g] = gi0[(size_t)t * 48 + g];
            float acc = sB[g];
            const float* wrow = &sW[g * 17];
#pragma unroll
            for (int k = 0; k < 16; ++k) acc += sH[k] * wrow[k];
            sGh[g] = acc;
        }
        __syncthreads();
        if (lane < 16) {
            float r  = fbrnn_sigmoid(sGi[lane] + sGh[lane]);
            float z  = fbrnn_sigmoid(sGi[16 + lane] + sGh[16 + lane]);
            float nn = tanhf(sGi[32 + lane] + r * sGh[32 + lane]);
            float hn = (1.0f - z) * nn + z * sH[lane];
            sNew[lane] = hn;
            sInp[lane] = hn;
        }
        __syncthreads();

        // -------- layers 1..5 --------
        for (int l = 1; l <= 5; ++l) {
            const float* wih = &sW[816 + (l - 1) * 1632];
            const float* whh = wih + 816;
            const float* bi  = &sB[48 + (l - 1) * 96];
            const float* bh  = bi + 48;
            for (int g = lane; g < 48; g += 32) {
                float ai = bi[g], ah = bh[g];
                const float* wi = &wih[g * 17];
                const float* wh = &whh[g * 17];
#pragma unroll
                for (int k = 0; k < 16; ++k) {
                    ai += sInp[k] * wi[k];
                    ah += sH[l * 16 + k] * wh[k];
                }
                sGi[g] = ai;
                sGh[g] = ah;
            }
            __syncthreads();
            if (lane < 16) {
                float r  = fbrnn_sigmoid(sGi[lane] + sGh[lane]);
                float z  = fbrnn_sigmoid(sGi[16 + lane] + sGh[16 + lane]);
                float nn = tanhf(sGi[32 + lane] + r * sGh[32 + lane]);
                float hn = (1.0f - z) * nn + z * sH[l * 16 + lane];
                sNew[l * 16 + lane] = hn;
                sInp[lane] = hn;
            }
            __syncthreads();
        }

        // -------- attention combine: layer i attends over newS[i..5] ------
        for (int i = 0; i < 6; ++i) {
            const int kcnt = 6 - i;
            float e[6];
            for (int j = 0; j < kcnt; ++j) {
                float p = 0.0f;
#pragma unroll
                for (int half = 0; half < 2; ++half) {
                    const int a = lane + half * 32;           // a in [0,64)
                    float acc = ba[i * 64 + a];
                    const float* s = &sNew[(i + j) * 16];
#pragma unroll
                    for (int c = 0; c < 16; ++c)
                        acc += s[c] * sWa[i * 1024 + c * 64 + a];
                    p += tanhf(acc) * va[i * 64 + a];
                }
                for (int off = 16; off > 0; off >>= 1)
                    p += __shfl_down(p, off, 32);
                e[j] = __shfl(p, 0, 32);
            }
            float mx = e[0];
            for (int j = 1; j < kcnt; ++j) mx = fmaxf(mx, e[j]);
            float den = 0.0f;
            for (int j = 0; j < kcnt; ++j) { e[j] = __expf(e[j] - mx); den += e[j]; }
            const float inv = 1.0f / den;
            if (lane < 16) {
                float acc = 0.0f;
                for (int j = 0; j < kcnt; ++j)
                    acc += e[j] * inv * sNew[(i + j) * 16 + lane];
                sH[i * 16 + lane] = acc;     // next-step hidden state
            }
            __syncthreads();
        }

        // -------- FC head on top = newS[5] (== h_next[5]) --------
        {
            float a1 = fc1b[lane];
#pragma unroll
            for (int c = 0; c < 16; ++c)
                a1 += sNew[80 + c] * fc1w[lane * 16 + c];
            float y = a1 * fc2w[lane];
            for (int off = 16; off > 0; off >>= 1)
                y += __shfl_down(y, off, 32);
            if (lane == 0) out[t] = y + fc2b[0];
        }
        __syncthreads();
    }
}

// ---------------------------------------------------------------------------
extern "C" void kernel_launch(void* const* d_in, const int* in_sizes, int n_in,
                              void* d_out, int out_size, void* d_ws, size_t ws_size,
                              hipStream_t stream)
{
    const float* batch = (const float*)d_in[0];
    const float* h0    = (const float*)d_in[1];
    const float* Wih0  = (const float*)d_in[2];
    const float* Whh0  = (const float*)d_in[3];
    const float* bih0  = (const float*)d_in[4];
    const float* bhh0  = (const float*)d_in[5];
    const float* Wih   = (const float*)d_in[6];
    const float* Whh   = (const float*)d_in[7];
    const float* bih   = (const float*)d_in[8];
    const float* bhh   = (const float*)d_in[9];
    const float* Wa    = (const float*)d_in[10];
    const float* ba    = (const float*)d_in[11];
    const float* va    = (const float*)d_in[12];
    const float* fc1w  = (const float*)d_in[13];
    const float* fc1b  = (const float*)d_in[14];
    const float* fc2w  = (const float*)d_in[15];
    const float* fc2b  = (const float*)d_in[16];
    float* out = (float*)d_out;

    const int T = 16384, F = 2048, G = 48;

    // scratch layout: [ gi0: T*48 f32 | wbf: 48*2048 bf16 ]
    float*  gi0 = (float*)d_ws;
    __bf16* wbf = (__bf16*)((char*)d_ws + (size_t)T * G * sizeof(float));

    fbrnn_cvt_w_bf16<<<(G * F + 255) / 256, 256, 0, stream>>>(Wih0, wbf, G * F);
    fbrnn_gemm_gi0<<<32, 256, 0, stream>>>(batch, wbf, bih0, gi0);
    fbrnn_scan<<<1, 32, 0, stream>>>(gi0, h0, Whh0, bhh0, Wih, Whh, bih, bhh,
                                     Wa, ba, va, fc1w, fc1b, fc2w, fc2b,
                                     out, T);
}
